// MultiHeadAttention_31129922962169
// MI455X (gfx1250) — compile-verified, matrix operand order checked by reading, not run on hardware
//
#include <hip/hip_runtime.h>

// ---------------- problem constants ----------------
#define B_  2
#define S_  2048
#define D_  1024
#define H_  16
#define HD_ 64

typedef unsigned short u16;
typedef __attribute__((ext_vector_type(16))) __bf16          v16bf;
typedef __attribute__((ext_vector_type(8)))  float           v8f;
typedef __attribute__((ext_vector_type(8)))  unsigned int    v8u;

// ---------------- helpers ----------------
__device__ __forceinline__ u16 f2bf(float f) {
  unsigned int u = __builtin_bit_cast(unsigned int, f);
  u += 0x7FFFu + ((u >> 16) & 1u);          // round-to-nearest-even
  return (u16)(u >> 16);
}

// Fragment whose 16 bf16 elements live at p[0..7] and p[16..23] (ISA 16-bit
// A/B per-lane layout). Two 16-byte vector loads (global_ or ds_ b128).
__device__ __forceinline__ v16bf load_frag_pair(const u16* p) {
  uint4 x = *(const uint4*)(p);
  uint4 y = *(const uint4*)(p + 16);
  v8u v;
  v[0] = x.x; v[1] = x.y; v[2] = x.z; v[3] = x.w;
  v[4] = y.x; v[5] = y.y; v[6] = y.z; v[7] = y.w;
  return __builtin_bit_cast(v16bf, v);
}

__device__ __forceinline__ v8f wmma_bf16(v16bf a, v16bf b, v8f c) {
  return __builtin_amdgcn_wmma_f32_16x16x32_bf16(
      /*neg_a=*/false, a, /*neg_b=*/false, b,
      /*c_mod=*/(short)0, c, /*reuse_a=*/false, /*reuse_b=*/false);
}

// LDS byte offset of a __shared__ object: flat LDS addresses keep the LDS
// offset in the low 32 bits (ISA 10.2 aperture mapping).
__device__ __forceinline__ unsigned lds_off(const void* p) {
  return (unsigned)(size_t)p;
}

// Async DMA: global -> LDS, 16 bytes per lane, tracked by ASYNCcnt.
__device__ __forceinline__ void async_load_b128(unsigned ldsoff, const void* gaddr) {
  asm volatile("global_load_async_to_lds_b128 %0, %1, off"
               :: "v"(ldsoff), "v"(gaddr) : "memory");
}
__device__ __forceinline__ void wait_asynccnt0() {
  asm volatile("s_wait_asynccnt 0" ::: "memory");
}

// ---------------- stage 0a: f32 -> bf16 (x) ----------------
__global__ void cvt_f32_bf16(const float* __restrict__ src, u16* __restrict__ dst, int n) {
  int i = blockIdx.x * blockDim.x + threadIdx.x;
  if (i < n) dst[i] = f2bf(src[i]);
}

// ---------------- stage 0b: f32 [K][N] -> bf16 transposed [N][K] ----------
__global__ void __launch_bounds__(256)
cvt_transpose_bf16(const float* __restrict__ src, u16* __restrict__ dst) {
  __shared__ u16 tile[32][33];
  const int bx = blockIdx.x * 32;          // N base
  const int by = blockIdx.y * 32;          // K base
  const int tx = threadIdx.x;              // 0..31
  const int ty = threadIdx.y;              // 0..7
#pragma unroll
  for (int i = ty; i < 32; i += 8)
    tile[i][tx] = f2bf(src[(size_t)(by + i) * D_ + bx + tx]);  // W[k][n]
  __syncthreads();
#pragma unroll
  for (int i = ty; i < 32; i += 8)
    dst[(size_t)(bx + i) * D_ + by + tx] = tile[tx][i];        // WT[n][k]
}

// ---------------- stages 1 & 4: bf16 GEMM via WMMA ----------------
// C[M,N] = A[M,K] @ B[K,N], B supplied transposed (BT[N][K]).
// Block = 256 threads = 8 waves (4 M x 2 N), wave tile 64x64 (4x4 WMMA:
// 16 WMMAs per 8 fragment loads), block tile 256x128.  A (256x32) and B
// (128x32) K-slices are async-DMA staged into double-buffered LDS.
//   mode 0: bf16 out [B,H,S,HD] (Q,K) | mode 2: bf16 out [B,H,HD,S] (V)
//   mode 1: f32 out + bias [M,N] (final projection)
__global__ void __launch_bounds__(256)
gemm_bf16(const u16* __restrict__ A, const u16* __restrict__ BT,
          u16* __restrict__ outBF, float* __restrict__ outF,
          const float* __restrict__ bias, int mode) {
  __shared__ alignas(16) u16 ldsA[2][256 * 32];   // 16KB per buffer
  __shared__ alignas(16) u16 ldsB[2][128 * 32];   //  8KB per buffer

  const int tid   = threadIdx.x;
  const int lane  = tid & 31;
  const int w     = tid >> 5;
  const int wm    = w >> 1;                // 0..3 -> M offset wm*64
  const int wn    = w & 1;                 // 0..1 -> N offset wn*64
  const int aBase = blockIdx.x * 256;
  const int bBase = blockIdx.y * 128;
  const int l15   = lane & 15;
  const int kb8   = (lane >> 4) << 3;      // K sub-base per half-wave
  const int rbase = (lane >> 4) << 3;      // C-frag row base per half-wave

  // staging: A = 1024 chunks (4/thread), B = 512 chunks (2/thread)
  const int rowc = tid >> 2;               // 0..63
  const int offc = (tid & 3) * 8;          // u16 offset within 32-elem row

  v8f acc[4][4] = {};

  auto stage = [&](int buf, int kb) {
#pragma unroll
    for (int c = 0; c < 4; ++c)            // A rows rowc, +64, +128, +192
      async_load_b128(lds_off(&ldsA[buf][(rowc + c * 64) * 32 + offc]),
                      A + (size_t)(aBase + rowc + c * 64) * D_ + kb + offc);
#pragma unroll
    for (int c = 0; c < 2; ++c)            // B rows rowc, +64
      async_load_b128(lds_off(&ldsB[buf][(rowc + c * 64) * 32 + offc]),
                      BT + (size_t)(bBase + rowc + c * 64) * D_ + kb + offc);
  };

  stage(0, 0);
  wait_asynccnt0();
  __syncthreads();

  int buf = 0;
  for (int kb = 0; kb < D_; kb += 32) {
    if (kb + 32 < D_) stage(buf ^ 1, kb + 32);   // overlap DMA with compute

    v16bf af[4], bfr[4];
#pragma unroll
    for (int m = 0; m < 4; ++m)
      af[m] = load_frag_pair(&ldsA[buf][(wm * 64 + m * 16 + l15) * 32 + kb8]);
#pragma unroll
    for (int n = 0; n < 4; ++n)
      bfr[n] = load_frag_pair(&ldsB[buf][(wn * 64 + n * 16 + l15) * 32 + kb8]);
#pragma unroll
    for (int m = 0; m < 4; ++m)
#pragma unroll
      for (int n = 0; n < 4; ++n)
        acc[m][n] = wmma_bf16(af[m], bfr[n], acc[m][n]);

    wait_asynccnt0();
    __syncthreads();
    buf ^= 1;
  }

#pragma unroll
  for (int am = 0; am < 4; ++am)
#pragma unroll
    for (int an = 0; an < 4; ++an)
#pragma unroll
      for (int r = 0; r < 8; ++r) {
        const int gRow = aBase + wm * 64 + am * 16 + rbase + r;
        const int gCol = bBase + wn * 64 + an * 16 + l15;
        const float v  = acc[am][an][r];
        if (mode == 1) {
          outF[(size_t)gRow * D_ + gCol] = v + bias[gCol];
        } else {
          const int bb = gRow >> 11;          // / S_
          const int s  = gRow & (S_ - 1);
          const int h  = gCol >> 6;           // / HD_
          const int hd = gCol & (HD_ - 1);
          if (mode == 0)
            outBF[(((size_t)bb * H_ + h) * S_ + s) * HD_ + hd] = f2bf(v);
          else   // mode 2: V stored transposed per head: [B,H,HD,S]
            outBF[(((size_t)bb * H_ + h) * HD_ + hd) * S_ + s] = f2bf(v);
        }
      }
}

// ---------------- stage 3: causal flash attention ----------------
// Grid (S/128, H, B), block 256 = 8 waves; each wave owns 16 query rows.
// K/V tiles for each 32-key block are async-DMA staged into double-buffered
// LDS shared by all 8 waves.  Trip count is uniform per block (fully-masked
// key blocks are exact no-ops in the online softmax), so barriers are legal.
__global__ void __launch_bounds__(256)
attn_kernel(const u16* __restrict__ Q, const u16* __restrict__ Kk,
            const u16* __restrict__ Vt, u16* __restrict__ ctx) {
  __shared__ alignas(16) u16 ldsK[2][32 * 64];   // [key][hd]  4KB per buffer
  __shared__ alignas(16) u16 ldsV[2][64 * 32];   // [hd][key]  4KB per buffer
  __shared__ alignas(16) u16 p_lds[8][16 * 32];  // per-wave P transpose

  const int tid  = threadIdx.x;
  const int lane = tid & 31;
  const int w    = tid >> 5;
  const int hi   = blockIdx.y;
  const int bi   = blockIdx.z;
  const int q0   = blockIdx.x * 128 + w * 16;

  const size_t hoff = ((size_t)(bi * H_ + hi)) * S_ * HD_;
  const u16* Qb = Q  + hoff;
  const u16* Kb = Kk + hoff;                // [S][HD] within head
  const u16* Vb = Vt + hoff;                // [HD][S] within head

  const int l15   = lane & 15;
  const int kb8   = (lane >> 4) << 3;
  const int rbase = (lane >> 4) << 3;

  // staging assignment (1 b128 chunk each for K and V per thread)
  const int krow = tid >> 3;                // 0..31 key row
  const int koff = (tid & 7) * 8;           // u16 offset in 64-elem row
  const int vrow = tid >> 2;                // 0..63 hd row
  const int voff = (tid & 3) * 8;           // u16 offset in 32-elem row

  auto stage = [&](int buf, int kc) {
    async_load_b128(lds_off(&ldsK[buf][krow * 64 + koff]),
                    Kb + (size_t)(kc + krow) * HD_ + koff);
    async_load_b128(lds_off(&ldsV[buf][vrow * 32 + voff]),
                    Vb + (size_t)vrow * S_ + kc + voff);
  };

  // Q A-fragments for HD slices [0,32) and [32,64): held across the loop
  const u16* qp = Qb + (size_t)(q0 + l15) * HD_ + kb8;
  const v16bf qa0 = load_frag_pair(qp);
  const v16bf qa1 = load_frag_pair(qp + 32);

  v8f oacc[4] = {};
  float m_i[8], l_i[8];
#pragma unroll
  for (int r = 0; r < 8; ++r) { m_i[r] = -1e30f; l_i[r] = 0.f; }

  const float scale = 0.125f;               // 1/sqrt(HD)
  const int nkb = blockIdx.x * 4 + 4;       // uniform per block (causal max)

  stage(0, 0);
  wait_asynccnt0();
  __syncthreads();

  int buf = 0;
  for (int kbi = 0; kbi < nkb; ++kbi) {
    const int kc = kbi * 32;
    if (kbi + 1 < nkb) stage(buf ^ 1, kc + 32);

    // ----- scores S(16x32) = Q @ K^T from the shared K tile
    v8f sc0 = {}, sc1 = {};
    {
      const u16* kp0 = &ldsK[buf][(l15) * 64 + kb8];
      sc0 = wmma_bf16(qa0, load_frag_pair(kp0), sc0);
      sc0 = wmma_bf16(qa1, load_frag_pair(kp0 + 32), sc0);
      const u16* kp1 = &ldsK[buf][(16 + l15) * 64 + kb8];
      sc1 = wmma_bf16(qa0, load_frag_pair(kp1), sc1);
      sc1 = wmma_bf16(qa1, load_frag_pair(kp1 + 32), sc1);
    }

    // ----- causal mask + online softmax (row stats via half-wave shuffles)
    float p0[8], p1[8];
#pragma unroll
    for (int r = 0; r < 8; ++r) {
      const int qrow = q0 + rbase + r;
      float s0 = sc0[r] * scale;
      float s1 = sc1[r] * scale;
      if (kc + l15 > qrow)      s0 = -1e30f;
      if (kc + 16 + l15 > qrow) s1 = -1e30f;
      float mx = fmaxf(s0, s1);
      mx = fmaxf(mx, __shfl_xor(mx, 1, 32));
      mx = fmaxf(mx, __shfl_xor(mx, 2, 32));
      mx = fmaxf(mx, __shfl_xor(mx, 4, 32));
      mx = fmaxf(mx, __shfl_xor(mx, 8, 32));
      const float mnew = fmaxf(m_i[r], mx);
      const float fac  = __expf(m_i[r] - mnew);
      const float e0   = __expf(s0 - mnew);
      const float e1   = __expf(s1 - mnew);
      float rs = e0 + e1;
      rs += __shfl_xor(rs, 1, 32);
      rs += __shfl_xor(rs, 2, 32);
      rs += __shfl_xor(rs, 4, 32);
      rs += __shfl_xor(rs, 8, 32);
      l_i[r] = l_i[r] * fac + rs;
      m_i[r] = mnew;
#pragma unroll
      for (int t = 0; t < 4; ++t) oacc[t][r] *= fac;
      p0[r] = e0;
      p1[r] = e1;
    }

    // ----- P: f32 C-layout -> bf16 A-layout via wave-local LDS transpose
#pragma unroll
    for (int r = 0; r < 8; ++r) {
      p_lds[w][(rbase + r) * 32 + l15]      = f2bf(p0[r]);
      p_lds[w][(rbase + r) * 32 + 16 + l15] = f2bf(p1[r]);
    }
    asm volatile("s_wait_dscnt 0" ::: "memory");  // wave-local visibility
    const v16bf pa = load_frag_pair(&p_lds[w][l15 * 32 + kb8]);

    // ----- O(16x64) += P(16x32) @ V(32x64) from the shared V tile
#pragma unroll
    for (int t = 0; t < 4; ++t) {
      v16bf vf = load_frag_pair(&ldsV[buf][(t * 16 + l15) * 32 + kb8]);
      oacc[t] = wmma_bf16(pa, vf, oacc[t]);
    }

    wait_asynccnt0();
    __syncthreads();
    buf ^= 1;
  }

  // ----- normalize and store ctx (bf16, merged heads [B,S,D])
  float inv[8];
#pragma unroll
  for (int r = 0; r < 8; ++r) inv[r] = 1.f / l_i[r];
#pragma unroll
  for (int t = 0; t < 4; ++t)
#pragma unroll
    for (int r = 0; r < 8; ++r) {
      const int q   = q0 + rbase + r;
      const int col = hi * HD_ + t * 16 + l15;
      ctx[((size_t)bi * S_ + q) * D_ + col] = f2bf(oacc[t][r] * inv[r]);
    }
}

// ---------------- launch ----------------
extern "C" void kernel_launch(void* const* d_in, const int* in_sizes, int n_in,
                              void* d_out, int out_size, void* d_ws, size_t ws_size,
                              hipStream_t stream) {
  const float* x  = (const float*)d_in[0];
  const float* Wq = (const float*)d_in[1];
  const float* Wk = (const float*)d_in[2];
  const float* Wv = (const float*)d_in[3];
  const float* Wo = (const float*)d_in[4];
  const float* bo = (const float*)d_in[5];
  float* out = (float*)d_out;

  const size_t NX = (size_t)B_ * S_ * D_;   // 4 Mi elems
  const size_t NW = (size_t)D_ * D_;        // 1 Mi elems

  u16* p    = (u16*)d_ws;
  u16* xb   = p; p += NX;
  u16* WqT  = p; p += NW;   // bf16, transposed [N][K]
  u16* WkT  = p; p += NW;
  u16* WvT  = p; p += NW;
  u16* WoT  = p; p += NW;
  u16* Qh   = p; p += NX;   // [B,H,S,HD] bf16
  u16* Kh   = p; p += NX;   // [B,H,S,HD] bf16
  u16* Vth  = p; p += NX;   // [B,H,HD,S] bf16 (transposed per head)
  u16* ctx  = p; p += NX;   // [B,S,D] bf16

  cvt_f32_bf16<<<(int)((NX + 255) / 256), 256, 0, stream>>>(x, xb, (int)NX);

  dim3 tgrid(D_ / 32, D_ / 32);
  dim3 tblk(32, 8);
  cvt_transpose_bf16<<<tgrid, tblk, 0, stream>>>(Wq, WqT);
  cvt_transpose_bf16<<<tgrid, tblk, 0, stream>>>(Wk, WkT);
  cvt_transpose_bf16<<<tgrid, tblk, 0, stream>>>(Wv, WvT);
  cvt_transpose_bf16<<<tgrid, tblk, 0, stream>>>(Wo, WoT);

  dim3 ggrid(B_ * S_ / 256, D_ / 128);      // (16, 8)
  gemm_bf16<<<ggrid, 256, 0, stream>>>(xb, WqT, Qh,  nullptr, nullptr, 0);
  gemm_bf16<<<ggrid, 256, 0, stream>>>(xb, WkT, Kh,  nullptr, nullptr, 0);
  gemm_bf16<<<ggrid, 256, 0, stream>>>(xb, WvT, Vth, nullptr, nullptr, 2);

  dim3 agrid(S_ / 128, H_, B_);             // (16, 16, 2)
  attn_kernel<<<agrid, 256, 0, stream>>>(Qh, Kh, Vth, ctx);

  gemm_bf16<<<ggrid, 256, 0, stream>>>(ctx, WoT, nullptr, out, bo, 1);
}
